// MoEFatehLayer_59528246722651
// MI455X (gfx1250) — compile-verified, matrix operand order checked by reading, not run on hardware
//
#include <hip/hip_runtime.h>
#include <hip/hip_bf16.h>
#include <math.h>

#define NTOK 8192      // B*S
#define DM   1024      // d_model
#define DFF  4096      // d_ff
#define NE   8         // experts
#define TILE_M 32      // tokens per workgroup tile
#define NTHR 512       // 16 wave32 per workgroup

typedef __attribute__((ext_vector_type(16))) __bf16 v16bf;
typedef __attribute__((ext_vector_type(8)))  __bf16 v8bf;
typedef __attribute__((ext_vector_type(8)))  float  v8f;

__device__ __forceinline__ __bf16 f2bf(float f) {
    unsigned u = __builtin_bit_cast(unsigned, f);
    u += 0x7fffu + ((u >> 16) & 1u);            // round-to-nearest-even
    unsigned short h = (unsigned short)(u >> 16);
    return __builtin_bit_cast(__bf16, h);
}

__device__ __forceinline__ v8f wmma_bf16(v16bf a, v16bf b, v8f c) {
    return __builtin_amdgcn_wmma_f32_16x16x32_bf16(false, a, false, b,
                                                   (short)0, c, false, false);
}

// ---------------- zero / init ----------------
__global__ void moe_zero_kernel(float* __restrict__ out, size_t n,
                                float* pm, float* hard, int* cnt) {
    size_t i = (size_t)blockIdx.x * blockDim.x + threadIdx.x;
    size_t stride = (size_t)gridDim.x * blockDim.x;
    for (size_t j = i; j < n; j += stride) out[j] = 0.0f;
    if (i < NE) { pm[i] = 0.0f; hard[i] = 0.0f; cnt[i] = 0; }
}

// ---------------- weight repack: f32 row-major -> bf16 WMMA-fragment-major ----
// Fragment (e, ntile, kb): 32 lanes x 16 bf16 contiguous (1 KB).
template<int NTILES, int KBT, int LDW>
__global__ void moe_repack_kernel(const float* __restrict__ W,
                                  __bf16* __restrict__ Wp) {
    size_t idx = (size_t)blockIdx.x * blockDim.x + threadIdx.x; // one 16-elem group
    int lane = (int)(idx & 31);
    size_t t = idx >> 5;
    int kb = (int)(t % KBT); t /= KBT;
    int nt = (int)(t % NTILES); t /= NTILES;
    int e  = (int)t;
    const float* src = W + (size_t)e * ((size_t)KBT * 32) * LDW;
    const int n  = nt * 16 + (lane & 15);
    const int kh = (lane >> 4) << 3;
    v16bf v;
    #pragma unroll
    for (int j = 0; j < 8; ++j) {
        v[j]     = f2bf(src[(size_t)(kb * 32 + kh + j)      * LDW + n]);
        v[j + 8] = f2bf(src[(size_t)(kb * 32 + kh + 16 + j) * LDW + n]);
    }
    *(v16bf*)(Wp + idx * 16) = v;
}

// ---------------- gating: one wave32 per token ----------------
__global__ void moe_gate_kernel(const float* __restrict__ x,
                                const float* __restrict__ Wg,
                                const float* __restrict__ bg,
                                int* __restrict__ cnt,
                                float* __restrict__ pm,
                                float* __restrict__ hard,
                                int* __restrict__ tok_list,
                                float* __restrict__ wgt_list) {
    const int wave = threadIdx.x >> 5;
    const int lane = threadIdx.x & 31;
    const int t = blockIdx.x * 8 + wave;   // 1024 blocks * 8 waves = 8192 tokens

    float a[NE];
    #pragma unroll
    for (int e = 0; e < NE; ++e) a[e] = 0.0f;

    const float* xr = x + (size_t)t * DM;
    for (int k = lane; k < DM; k += 32) {
        float xv = xr[k];
        const float* wr = Wg + (size_t)k * NE;
        #pragma unroll
        for (int e = 0; e < NE; ++e) a[e] += xv * wr[e];
    }
    #pragma unroll
    for (int e = 0; e < NE; ++e) {
        #pragma unroll
        for (int off = 16; off > 0; off >>= 1)
            a[e] += __shfl_xor(a[e], off, 32);
    }

    float l[NE], p[NE];
    float mx = -1e30f;
    #pragma unroll
    for (int e = 0; e < NE; ++e) { l[e] = a[e] + bg[e]; mx = fmaxf(mx, l[e]); }
    float s = 0.0f;
    #pragma unroll
    for (int e = 0; e < NE; ++e) { p[e] = __expf(l[e] - mx); s += p[e]; }
    float inv = 1.0f / s;
    #pragma unroll
    for (int e = 0; e < NE; ++e) p[e] *= inv;

    int i1 = 0;
    #pragma unroll
    for (int e = 1; e < NE; ++e) if (p[e] > p[i1]) i1 = e;
    int i2 = (i1 == 0) ? 1 : 0;
    #pragma unroll
    for (int e = 0; e < NE; ++e) if (e != i1 && p[e] > p[i2]) i2 = e;
    float wsum = p[i1] + p[i2];
    float w1 = p[i1] / wsum, w2 = p[i2] / wsum;

    if (lane == 0) {
        #pragma unroll
        for (int e = 0; e < NE; ++e) atomicAdd(&pm[e], p[e]);
        atomicAdd(&hard[i1], 1.0f);   // argmax == top-1
        int pos = atomicAdd(&cnt[i1], 1);
        tok_list[i1 * NTOK + pos] = t; wgt_list[i1 * NTOK + pos] = w1;
        pos = atomicAdd(&cnt[i2], 1);
        tok_list[i2 * NTOK + pos] = t; wgt_list[i2 * NTOK + pos] = w2;
    }
}

// ---------------- aux loss finalize ----------------
__global__ void moe_aux_kernel(const float* __restrict__ pm,
                               const float* __restrict__ hard,
                               float* __restrict__ out) {
    int l = threadIdx.x;
    float v = (l < NE) ? pm[l] * hard[l] : 0.0f;
    #pragma unroll
    for (int off = 16; off > 0; off >>= 1) v += __shfl_xor(v, off, 32);
    if (l == 0)
        out[(size_t)NTOK * DM] = (float)NE * v / ((float)NTOK * (float)NTOK);
}

// ---------------- expert FFN: 32-token tile, WMMA bf16, repacked weights ----
// block = 512 threads (16 wave32); grid = (256 tiles, 8 experts)
// Each wave owns 64 columns (4 N-tiles) -> 64 persistent acc VGPRs (no spills).
__global__ void __launch_bounds__(NTHR, 1)
moe_expert_kernel(const float* __restrict__ x,
                  const __bf16* __restrict__ W1p,
                  const float* __restrict__ b1,
                  const __bf16* __restrict__ W2p,
                  const float* __restrict__ b2,
                  const int* __restrict__ cnt,
                  const int* __restrict__ tok_list,
                  const float* __restrict__ wgt_list,
                  float* __restrict__ out) {
    const int e = blockIdx.y;
    const int tile = blockIdx.x;
    const int count = cnt[e];
    if (tile * TILE_M >= count) return;

    extern __shared__ char smem[];
    __bf16* Xls = (__bf16*)smem;                 // [32][1024] bf16 = 64 KB
    __bf16* Hls = Xls + TILE_M * DM;             // [32][1024] bf16 = 64 KB (chunk)
    int*    toks = (int*)(Hls + TILE_M * 1024);  // [32]
    float*  wts  = (float*)(toks + TILE_M);      // [32]

    const int tid  = threadIdx.x;
    const int wave = tid >> 5;                   // 0..15
    const int lane = tid & 31;

    if (tid < TILE_M) {
        int idx = tile * TILE_M + tid;
        int t = 0; float w = 0.0f;
        if (idx < count) { t = tok_list[e * NTOK + idx]; w = wgt_list[e * NTOK + idx]; }
        toks[tid] = t; wts[tid] = w;    // padded rows: weight 0 -> contributes 0
    }
    __syncthreads();

    // stage X tile -> LDS as bf16 (coalesced along k)
    for (int i = tid; i < TILE_M * DM; i += NTHR) {
        int m = i >> 10, k = i & (DM - 1);
        Xls[i] = f2bf(x[(size_t)toks[m] * DM + k]);
    }
    __syncthreads();

    const float* b1e = b1 + (size_t)e * DFF;
    const float* b2e = b2 + (size_t)e * DM;

    const int row   = lane & 15;            // A-matrix M row within row-tile
    const int kh8   = (lane >> 4) << 3;     // K sub-block select per lane half
    const int nlane = lane & 15;            // C/D column within tile
    const int mh    = (lane >> 4) << 3;     // C/D row offset per lane half

    v8f acc[2][4];
    #pragma unroll
    for (int mt = 0; mt < 2; ++mt)
        #pragma unroll
        for (int i = 0; i < 4; ++i) acc[mt][i] = (v8f){};

    for (int c = 0; c < 4; ++c) {           // d_ff chunks of 1024
        // ---- GEMM1: H_chunk[32,1024] = gelu(X @ W1[:, chunk] + b1) ----
        #pragma unroll
        for (int nt = 0; nt < 4; ++nt) {
            const int ntg = c * 64 + wave * 4 + nt;              // global d_ff tile
            const __bf16* bfrag = W1p + ((size_t)(e * 256 + ntg) * 32) * 512
                                      + (size_t)lane * 16;
            v8f h0 = (v8f){}, h1 = (v8f){};
            for (int kb = 0; kb < 32; ++kb) {
                v16bf Bv = *(const v16bf*)(bfrag + (size_t)kb * 512);
                const int ko = kb * 32 + kh8;
                union { v16bf v; v8bf half[2]; } A0, A1;
                A0.half[0] = *(const v8bf*)&Xls[row * DM + ko];
                A0.half[1] = *(const v8bf*)&Xls[row * DM + ko + 16];
                A1.half[0] = *(const v8bf*)&Xls[(16 + row) * DM + ko];
                A1.half[1] = *(const v8bf*)&Xls[(16 + row) * DM + ko + 16];
                h0 = wmma_bf16(A0.v, Bv, h0);
                h1 = wmma_bf16(A1.v, Bv, h1);
            }
            const int colloc = (wave * 4 + nt) * 16;             // local chunk col
            const float bb = b1e[c * 1024 + colloc + nlane];
            #pragma unroll
            for (int r = 0; r < 8; ++r) {
                float v0 = h0[r] + bb;
                float v1 = h1[r] + bb;
                v0 = 0.5f * v0 * (1.0f + erff(v0 * 0.70710678118654752f));
                v1 = 0.5f * v1 * (1.0f + erff(v1 * 0.70710678118654752f));
                Hls[(r + mh) * 1024 + colloc + nlane]      = f2bf(v0);
                Hls[(16 + r + mh) * 1024 + colloc + nlane] = f2bf(v1);
            }
        }
        __syncthreads();

        // ---- GEMM2: acc += H_chunk @ W2[chunk, :]  (kb-outer: A reused 4x) ----
        {
            const __bf16* bbase = W2p + ((size_t)((e * 64 + wave * 4) * 128 + c * 32)) * 512
                                      + (size_t)lane * 16;
            for (int kb = 0; kb < 32; ++kb) {
                const int ko = kb * 32 + kh8;
                union { v16bf v; v8bf half[2]; } A0, A1;
                A0.half[0] = *(const v8bf*)&Hls[row * 1024 + ko];
                A0.half[1] = *(const v8bf*)&Hls[row * 1024 + ko + 16];
                A1.half[0] = *(const v8bf*)&Hls[(16 + row) * 1024 + ko];
                A1.half[1] = *(const v8bf*)&Hls[(16 + row) * 1024 + ko + 16];
                #pragma unroll
                for (int nt = 0; nt < 4; ++nt) {
                    v16bf Bv = *(const v16bf*)(bbase + ((size_t)nt * 128 * 512)
                                                     + (size_t)kb * 512);
                    acc[0][nt] = wmma_bf16(A0.v, Bv, acc[0][nt]);
                    acc[1][nt] = wmma_bf16(A1.v, Bv, acc[1][nt]);
                }
            }
        }
        __syncthreads();   // before next chunk overwrites Hls
    }

    // ---- epilogue: scale by routing weight, scatter-add ----
    #pragma unroll
    for (int nt = 0; nt < 4; ++nt) {
        const int ocol = (wave * 4 + nt) * 16 + nlane;
        const float bb = b2e[ocol];
        #pragma unroll
        for (int mt = 0; mt < 2; ++mt) {
            #pragma unroll
            for (int r = 0; r < 8; ++r) {
                const int m = mt * 16 + r + mh;
                const float v = wts[m] * (acc[mt][nt][r] + bb);
                atomicAdd(&out[(size_t)toks[m] * DM + ocol], v);
            }
        }
    }
}

// ---------------- host launcher ----------------
extern "C" void kernel_launch(void* const* d_in, const int* in_sizes, int n_in,
                              void* d_out, int out_size, void* d_ws, size_t ws_size,
                              hipStream_t stream) {
    const float* x  = (const float*)d_in[0];
    const float* Wg = (const float*)d_in[1];
    const float* bg = (const float*)d_in[2];
    const float* W1 = (const float*)d_in[3];
    const float* b1 = (const float*)d_in[4];
    const float* W2 = (const float*)d_in[5];
    const float* b2 = (const float*)d_in[6];
    float* out = (float*)d_out;

    char* ws = (char*)d_ws;
    int*   cnt      = (int*)ws;                          // [8]
    float* pm       = (float*)(ws + 32);                 // [8]
    float* hard     = (float*)(ws + 64);                 // [8]
    int*   tok_list = (int*)(ws + 128);                  // [8][8192]
    float* wgt_list = (float*)(ws + 128 + (size_t)NE * NTOK * sizeof(int));
    // repacked bf16 weights (1 MB aligned): 64 MB each
    __bf16* W1p = (__bf16*)(ws + (1u << 20));
    __bf16* W2p = (__bf16*)(ws + (1u << 20) + (size_t)NE * DM * DFF * sizeof(__bf16));

    moe_zero_kernel<<<512, 256, 0, stream>>>(out, (size_t)NTOK * DM + 1, pm, hard, cnt);

    // repack: NE * ntiles * kbt * 32 lane-groups, 256 threads/block
    moe_repack_kernel<DFF / 16, DM / 32, DFF><<<NE * (DFF / 16) * (DM / 32) * 32 / 256,
                                                256, 0, stream>>>(W1, W1p);
    moe_repack_kernel<DM / 16, DFF / 32, DM><<<NE * (DM / 16) * (DFF / 32) * 32 / 256,
                                               256, 0, stream>>>(W2, W2p);

    moe_gate_kernel<<<NTOK / 8, 256, 0, stream>>>(x, Wg, bg, cnt, pm, hard,
                                                  tok_list, wgt_list);
    moe_aux_kernel<<<1, 32, 0, stream>>>(pm, hard, out);

    size_t smem = (size_t)TILE_M * DM * sizeof(__bf16)      // X tile
                + (size_t)TILE_M * 1024 * sizeof(__bf16)    // H chunk
                + TILE_M * (sizeof(int) + sizeof(float));
    moe_expert_kernel<<<dim3(NTOK / TILE_M, NE), NTHR, smem, stream>>>(
        x, W1p, b1, W2p, b2, cnt, tok_list, wgt_list, out);
}
